// MultiHeadSelfAttention_1357209665786
// MI455X (gfx1250) — matrix operand order repaired and checked by reading, not compiled, not run on hardware
//
#include <hip/hip_runtime.h>
#include <hip/hip_bf16.h>
#include <cstdint>

// ---------------------------------------------------------------------------
// MI455X (gfx1250) multi-head self-attention with relative position bias.
// All matrix math on v_wmma_f32_16x16x32_f16 (f16 operands, f32 accumulate).
// Flash-style online softmax; rel-bias computed as Q @ rel^T over a 527-wide
// window per 16-query tile (WMMA) and gathered with the skew trick from LDS.
// ---------------------------------------------------------------------------

typedef _Float16 v16h __attribute__((ext_vector_type(16)));
typedef _Float16 v8h  __attribute__((ext_vector_type(8)));
typedef float    v8f  __attribute__((ext_vector_type(8)));

#define D_MODEL 512
#define SEQ     512
#define NB      8
#define NH      8
#define DKH     64
#define RELROWS 1023

static __device__ __forceinline__ v8f wmma_f16(v16h a, v16h b, v8f c) {
  // 8 args: (neg_a, A, neg_b, B, c_mod, C, reuse_a, reuse_b)
  return __builtin_amdgcn_wmma_f32_16x16x32_f16(false, a, false, b, (short)0, c,
                                                false, false);
}

// A-matrix 16x32 f16 fragment: lane L holds row M=L%16;
// elems 0..15 -> K=(L/16)*16 + i (contiguous, matches B fragments).
static __device__ __forceinline__ v16h load_a_frag(const _Float16* __restrict__ base,
                                                   int row0, int col0, int ld) {
  const int lane = threadIdx.x & 31;
  const int r = lane & 15, hf = lane >> 4;
  const _Float16* p = base + (size_t)(row0 + r) * ld + col0 + hf * 16;
  v8h lo = *(const v8h*)p;
  v8h hi = *(const v8h*)(p + 8);
  v16h out;
#pragma unroll
  for (int i = 0; i < 8; ++i) { out[i] = lo[i]; out[i + 8] = hi[i]; }
  return out;
}

// B-matrix 32x16 f16 fragment: lane L holds column N=L%16; lanes 0-15 hold
// K=0..15, lanes 16-31 hold K=16..31 (contiguous). Storage is column-major B
// i.e. row-major B^T with leading dim ld: 16 contiguous halves per lane.
static __device__ __forceinline__ v16h load_b_frag(const _Float16* __restrict__ base,
                                                   int nrow0, int col0, int ld) {
  const int lane = threadIdx.x & 31;
  const int n = lane & 15, hf = lane >> 4;
  const _Float16* p = base + (size_t)(nrow0 + n) * ld + col0 + hf * 16;
  v8h lo = *(const v8h*)p;
  v8h hi = *(const v8h*)(p + 8);
  v16h out;
#pragma unroll
  for (int i = 0; i < 8; ++i) { out[i] = lo[i]; out[i + 8] = hi[i]; }
  return out;
}

// ---------------------------------------------------------------------------
// f32 -> f16 conversion
// ---------------------------------------------------------------------------
__global__ void cvt_f32_f16(const float* __restrict__ s, _Float16* __restrict__ d,
                            int n) {
  int i = blockIdx.x * blockDim.x + threadIdx.x;
  int stride = gridDim.x * blockDim.x;
  for (; i < n; i += stride) d[i] = (_Float16)s[i];
}

// ---------------------------------------------------------------------------
// Fused QKV projection: C = X(4096x512) @ W^T + b for W in {Wq,Wk,Wv}.
// Q,K stored per-head row-major (bh, s, d); V stored per-head TRANSPOSED
// (bh, d, s) so attn@V B-fragments are contiguous.
// One wave computes a 16x64 output tile (64 WMMAs per wave).
// ---------------------------------------------------------------------------
__global__ __launch_bounds__(32)
void qkv_gemm(const _Float16* __restrict__ xh,
              const _Float16* __restrict__ wqh, const _Float16* __restrict__ wkh,
              const _Float16* __restrict__ wvh,
              const float* __restrict__ bq, const float* __restrict__ bk,
              const float* __restrict__ bv,
              _Float16* __restrict__ qh, _Float16* __restrict__ kh,
              _Float16* __restrict__ vT) {
  const int row0 = blockIdx.x * 16;   // 0..4095  (b*512 + s)
  const int n0   = blockIdx.y * 64;   // 0..511   (h*64 + d), head-aligned
  const int z    = blockIdx.z;        // 0=Q 1=K 2=V
  const _Float16* W = (z == 0) ? wqh : ((z == 1) ? wkh : wvh);
  const float* bias = (z == 0) ? bq  : ((z == 1) ? bk  : bv);

  v8f acc[4] = {};
  for (int kk = 0; kk < D_MODEL; kk += 32) {
    v16h a = load_a_frag(xh, row0, kk, D_MODEL);
#pragma unroll
    for (int f = 0; f < 4; ++f) {
      // B[k][n] = W[n][k] : W row-major == B column-major, contiguous in k.
      v16h bf = load_b_frag(W, n0 + f * 16, kk, D_MODEL);
      acc[f] = wmma_f16(a, bf, acc[f]);
    }
  }
  const int lane = threadIdx.x & 31;
  const int cn = lane & 15, chf = lane >> 4;
  const int hh = n0 >> 6;
#pragma unroll
  for (int f = 0; f < 4; ++f) {
#pragma unroll
    for (int r = 0; r < 8; ++r) {
      const int m = r + 8 * chf;
      const int grow = row0 + m;
      const int bb = grow >> 9, ss = grow & 511;
      const int j = n0 + f * 16 + cn;
      const int d = f * 16 + cn;                 // j - hh*64
      const float val = acc[f][r] + bias[j];
      if (z == 2) {
        vT[(((size_t)(bb * NH + hh)) * DKH + d) * SEQ + ss] = (_Float16)val;
      } else {
        _Float16* dst = (z == 0) ? qh : kh;
        dst[(((size_t)(bb * NH + hh)) * SEQ + ss) * DKH + d] = (_Float16)val;
      }
    }
  }
}

// ---------------------------------------------------------------------------
// Flash attention + relative bias. One wave per (b,h,q-tile of 16).
// ---------------------------------------------------------------------------
__global__ __launch_bounds__(32)
void attn_kernel(const _Float16* __restrict__ qh, const _Float16* __restrict__ kh,
                 const _Float16* __restrict__ vT, const _Float16* __restrict__ relh,
                 _Float16* __restrict__ attn_out) {
  __shared__ float    G[16][528];   // Q_tile @ rel^T over the 527-wide window
  __shared__ _Float16 P[16][32];    // exp'd probabilities, C->A layout bounce

  const int qt = blockIdx.x;        // 0..31
  const int bh = blockIdx.y;        // 0..63
  const int bb = bh >> 3, hh = bh & 7;
  const int q0 = qt * 16;
  const int lane = threadIdx.x & 31;
  const int cn = lane & 15, chf = lane >> 4;

  const _Float16* qbase = qh + (size_t)bh * SEQ * DKH;
  const _Float16* kbase = kh + (size_t)bh * SEQ * DKH;
  const _Float16* vbase = vT + (size_t)bh * DKH * SEQ;

  const v16h aq0 = load_a_frag(qbase, q0, 0, DKH);
  const v16h aq1 = load_a_frag(qbase, q0, 32, DKH);

  // --- Phase 1: G[m][c] = dot(q[q0+m], rel[jbase+c, hh*64:hh*64+64]) ---
  const int jbase = 496 - q0;       // q0 <= 496 so jbase >= 0
  for (int jt = 0; jt < 33; ++jt) {
    int j = jbase + jt * 16 + cn;
    if (j > RELROWS - 1) j = RELROWS - 1;       // clamp; column 527 never read
    const _Float16* pr = relh + (size_t)j * D_MODEL + hh * 64;
    v16h b0, b1;
    {
      const _Float16* p0 = pr + chf * 16;
      v8h lo = *(const v8h*)p0, hi = *(const v8h*)(p0 + 8);
#pragma unroll
      for (int i = 0; i < 8; ++i) { b0[i] = lo[i]; b0[i + 8] = hi[i]; }
      const _Float16* p1 = pr + 32 + chf * 16;
      lo = *(const v8h*)p1; hi = *(const v8h*)(p1 + 8);
#pragma unroll
      for (int i = 0; i < 8; ++i) { b1[i] = lo[i]; b1[i + 8] = hi[i]; }
    }
    v8f c = {};
    c = wmma_f16(aq0, b0, c);
    c = wmma_f16(aq1, b1, c);
#pragma unroll
    for (int r = 0; r < 8; ++r) G[r + 8 * chf][jt * 16 + cn] = c[r];
  }
  __syncthreads();

  // --- Phase 2: flash loop over key pairs (32 keys at a time) ---
  float mrow[8], lrow[8];
#pragma unroll
  for (int r = 0; r < 8; ++r) { mrow[r] = -1e30f; lrow[r] = 0.0f; }
  v8f O[4] = {};

  for (int pt = 0; pt < 16; ++pt) {
    float sc[2][8];
#pragma unroll
    for (int sub = 0; sub < 2; ++sub) {
      const int kt = pt * 2 + sub;
      v16h bk0 = load_b_frag(kbase, kt * 16, 0, DKH);   // B[d][kp]=k[kp][d]
      v16h bk1 = load_b_frag(kbase, kt * 16, 32, DKH);
      v8f a = {};
      a = wmma_f16(aq0, bk0, a);
      a = wmma_f16(aq1, bk1, a);
#pragma unroll
      for (int r = 0; r < 8; ++r) {
        const int m = r + 8 * chf;
        const int gi = kt * 16 + cn - m + 15;           // skew gather, 0..526
        sc[sub][r] = a[r] * 0.125f + G[m][gi];
      }
    }
    // row max across the 16 lanes of each half (n dimension)
    float mt[8];
#pragma unroll
    for (int r = 0; r < 8; ++r) mt[r] = fmaxf(sc[0][r], sc[1][r]);
#pragma unroll
    for (int off = 1; off < 16; off <<= 1)
#pragma unroll
      for (int r = 0; r < 8; ++r)
        mt[r] = fmaxf(mt[r], __shfl_xor(mt[r], off, 32));

    float p0[8], p1[8], rs[8], scl[8];
#pragma unroll
    for (int r = 0; r < 8; ++r) {
      const float mn = fmaxf(mrow[r], mt[r]);
      scl[r] = __expf(mrow[r] - mn);
      p0[r] = __expf(sc[0][r] - mn);
      p1[r] = __expf(sc[1][r] - mn);
      rs[r] = p0[r] + p1[r];
      mrow[r] = mn;
      lrow[r] *= scl[r];
    }
#pragma unroll
    for (int off = 1; off < 16; off <<= 1)
#pragma unroll
      for (int r = 0; r < 8; ++r) rs[r] += __shfl_xor(rs[r], off, 32);
#pragma unroll
    for (int r = 0; r < 8; ++r) lrow[r] += rs[r];
#pragma unroll
    for (int f = 0; f < 4; ++f)
#pragma unroll
      for (int r = 0; r < 8; ++r) O[f][r] *= scl[r];

    // C-layout -> A-layout via LDS bounce
#pragma unroll
    for (int r = 0; r < 8; ++r) {
      const int m = r + 8 * chf;
      P[m][cn]      = (_Float16)p0[r];
      P[m][16 + cn] = (_Float16)p1[r];
    }
    __syncthreads();
    v16h ap;
    {
      const int rr = lane & 15, hf = lane >> 4;
#pragma unroll
      for (int i = 0; i < 8; ++i) {
        ap[i]     = P[rr][hf * 16 + i];
        ap[i + 8] = P[rr][hf * 16 + 8 + i];
      }
    }
    // O += P(16x32) @ V(32x64); V transposed so B frags are contiguous
#pragma unroll
    for (int f = 0; f < 4; ++f) {
      v16h bv = load_b_frag(vbase, f * 16, pt * 32, SEQ);
      O[f] = wmma_f16(ap, bv, O[f]);
    }
    __syncthreads();
  }

  // --- finalize: O / l, store f16 (b, s, h*64+d) for out projection ---
  float rinv[8];
#pragma unroll
  for (int r = 0; r < 8; ++r) rinv[r] = 1.0f / lrow[r];
#pragma unroll
  for (int f = 0; f < 4; ++f)
#pragma unroll
    for (int r = 0; r < 8; ++r) {
      const int m = r + 8 * chf;
      const float val = O[f][r] * rinv[r];
      attn_out[((size_t)(bb * SEQ + q0 + m)) * D_MODEL + hh * 64 + f * 16 + cn] =
          (_Float16)val;
    }
}

// ---------------------------------------------------------------------------
// Output projection: out = attn(4096x512 f16) @ Wo^T + bo  (f32 out)
// ---------------------------------------------------------------------------
__global__ __launch_bounds__(32)
void out_proj(const _Float16* __restrict__ ah, const _Float16* __restrict__ woh,
              const float* __restrict__ bo, float* __restrict__ out) {
  const int row0 = blockIdx.x * 16;
  const int n0   = blockIdx.y * 64;
  v8f acc[4] = {};
  for (int kk = 0; kk < D_MODEL; kk += 32) {
    v16h a = load_a_frag(ah, row0, kk, D_MODEL);
#pragma unroll
    for (int f = 0; f < 4; ++f) {
      v16h bf = load_b_frag(woh, n0 + f * 16, kk, D_MODEL);
      acc[f] = wmma_f16(a, bf, acc[f]);
    }
  }
  const int lane = threadIdx.x & 31;
  const int cn = lane & 15, chf = lane >> 4;
#pragma unroll
  for (int f = 0; f < 4; ++f)
#pragma unroll
    for (int r = 0; r < 8; ++r) {
      const int m = r + 8 * chf;
      const int j = n0 + f * 16 + cn;
      out[(size_t)(row0 + m) * D_MODEL + j] = acc[f][r] + bo[j];
    }
}

// ---------------------------------------------------------------------------
// Workspace layout (in halves)
// ---------------------------------------------------------------------------
static const size_t OFF_X   = 0;
static const size_t OFF_WQ  = OFF_X  + (size_t)4096 * 512;
static const size_t OFF_WK  = OFF_WQ + (size_t)512 * 512;
static const size_t OFF_WV  = OFF_WK + (size_t)512 * 512;
static const size_t OFF_WO  = OFF_WV + (size_t)512 * 512;
static const size_t OFF_REL = OFF_WO + (size_t)512 * 512;
static const size_t OFF_Q   = OFF_REL + (size_t)RELROWS * 512;
static const size_t OFF_K   = OFF_Q  + (size_t)64 * 512 * 64;
static const size_t OFF_V   = OFF_K  + (size_t)64 * 512 * 64;
static const size_t OFF_AT  = OFF_V  + (size_t)64 * 512 * 64;

extern "C" void kernel_launch(void* const* d_in, const int* in_sizes, int n_in,
                              void* d_out, int out_size, void* d_ws, size_t ws_size,
                              hipStream_t stream) {
  (void)in_sizes; (void)n_in; (void)out_size; (void)ws_size;
  const float* x   = (const float*)d_in[0];
  const float* Wq  = (const float*)d_in[1];
  const float* bq  = (const float*)d_in[2];
  const float* Wk  = (const float*)d_in[3];
  const float* bk  = (const float*)d_in[4];
  const float* Wv  = (const float*)d_in[5];
  const float* bv  = (const float*)d_in[6];
  const float* Wo  = (const float*)d_in[7];
  const float* bo  = (const float*)d_in[8];
  const float* rel = (const float*)d_in[9];
  float* out = (float*)d_out;
  _Float16* ws = (_Float16*)d_ws;

  _Float16* xh   = ws + OFF_X;
  _Float16* wqh  = ws + OFF_WQ;
  _Float16* wkh  = ws + OFF_WK;
  _Float16* wvh  = ws + OFF_WV;
  _Float16* woh  = ws + OFF_WO;
  _Float16* relh = ws + OFF_REL;
  _Float16* qhp  = ws + OFF_Q;
  _Float16* khp  = ws + OFF_K;
  _Float16* vTp  = ws + OFF_V;
  _Float16* ath  = ws + OFF_AT;

  cvt_f32_f16<<<1024, 256, 0, stream>>>(x,   xh,   4096 * 512);
  cvt_f32_f16<<<256,  256, 0, stream>>>(Wq,  wqh,  512 * 512);
  cvt_f32_f16<<<256,  256, 0, stream>>>(Wk,  wkh,  512 * 512);
  cvt_f32_f16<<<256,  256, 0, stream>>>(Wv,  wvh,  512 * 512);
  cvt_f32_f16<<<256,  256, 0, stream>>>(Wo,  woh,  512 * 512);
  cvt_f32_f16<<<512,  256, 0, stream>>>(rel, relh, RELROWS * 512);

  qkv_gemm<<<dim3(256, 8, 3), 32, 0, stream>>>(xh, wqh, wkh, wvh, bq, bk, bv,
                                               qhp, khp, vTp);
  attn_kernel<<<dim3(32, 64), 32, 0, stream>>>(qhp, khp, vTp, relh, ath);
  out_proj<<<dim3(256, 8), 32, 0, stream>>>(ath, woh, bo, out);
}
